// CrossNetMix_44006234914963
// MI455X (gfx1250) — compile-verified, matrix operand order checked by reading, not run on hardware
//
#include <hip/hip_runtime.h>
#include <hip/hip_bf16.h>

// ---------------------------------------------------------------------------
// CrossNetMix fused kernel for gfx1250 (MI455X), bf16 WMMA path.
// B=16384, D=1024, R=64, E=4, L=3. Row-independent recurrence -> single fused
// kernel; weights repacked once to bf16 in WMMA B-fragment layout in d_ws.
// ---------------------------------------------------------------------------

typedef __attribute__((ext_vector_type(16))) __bf16 v16bf;
typedef __attribute__((ext_vector_type(8)))  float  v8f;

#define LAYERS 3
#define BATCH  16384
#define DIM    1024
#define RANK   64
#define NEXP   4

// Packed-weight region layout (in bf16 elements). One fragment = 32x16 bf16
// B-tile = 512 elements = 1KB, stored as lane-major (lane*16 + q).
#define FRAG_E        512
#define G_FRAGS       32            // gates: K=1024 -> 32 k-chunks, one N-tile
#define V_FRAGS_PER_L (32*16)       // K=1024 (32 kc) x N=256 (16 nt)
#define C_FRAGS_PER_L (4*2*4)       // E x kc(2) x nt(4)
#define U_FRAGS_PER_L (8*64)        // K=256 (8 kc) x N=1024 (64 nt)
#define GOFF 0
#define VOFF (G_FRAGS*FRAG_E)
#define COFF (VOFF + LAYERS*V_FRAGS_PER_L*FRAG_E)
#define UOFF (COFF + LAYERS*C_FRAGS_PER_L*FRAG_E)
#define TOTE (UOFF + LAYERS*U_FRAGS_PER_L*FRAG_E)

__device__ __forceinline__ unsigned short f2bf(float x) {
  // round-to-nearest-even fp32 -> bf16
  unsigned int u = __float_as_uint(x);
  u += 0x7FFFu + ((u >> 16) & 1u);
  return (unsigned short)(u >> 16);
}

union FragU {
  uint4 u4[2];
  unsigned short us[16];
  v16bf v;
};

// B fragment: pre-packed, lane's 16 halfs are contiguous (32B per lane).
__device__ __forceinline__ v16bf load_bfrag(const unsigned short* frag, int lane) {
  FragU f;
  const uint4* p = reinterpret_cast<const uint4*>(frag) + (lane << 1);
  f.u4[0] = p[0];
  f.u4[1] = p[1];
  return f.v;
}

// A fragment (16x32 bf16) from a row-major bf16 LDS buffer.
// ISA layout: lane = m + 16h; halfs q0..7 -> k = kb+8h+q ; q8..15 -> k = kb+16+8h+(q-8)
__device__ __forceinline__ v16bf afrag_bf16(const unsigned short* base, int rowstride,
                                            int kb, int lane) {
  int m = lane & 15, h = lane >> 4;
  const unsigned short* r = base + m * rowstride + kb + 8 * h;
  FragU f;
  f.u4[0] = *reinterpret_cast<const uint4*>(r);
  f.u4[1] = *reinterpret_cast<const uint4*>(r + 16);
  return f.v;
}

// A fragment from a row-major fp32 LDS buffer (on-the-fly bf16 convert).
__device__ __forceinline__ v16bf afrag_f32(const float* base, int rowstride,
                                           int kb, int lane) {
  int m = lane & 15, h = lane >> 4;
  const float* r = base + m * rowstride + kb + 8 * h;
  float4 f0 = *reinterpret_cast<const float4*>(r);
  float4 f1 = *reinterpret_cast<const float4*>(r + 4);
  float4 f2 = *reinterpret_cast<const float4*>(r + 16);
  float4 f3 = *reinterpret_cast<const float4*>(r + 20);
  FragU f;
  f.us[0]  = f2bf(f0.x); f.us[1]  = f2bf(f0.y); f.us[2]  = f2bf(f0.z); f.us[3]  = f2bf(f0.w);
  f.us[4]  = f2bf(f1.x); f.us[5]  = f2bf(f1.y); f.us[6]  = f2bf(f1.z); f.us[7]  = f2bf(f1.w);
  f.us[8]  = f2bf(f2.x); f.us[9]  = f2bf(f2.y); f.us[10] = f2bf(f2.z); f.us[11] = f2bf(f2.w);
  f.us[12] = f2bf(f3.x); f.us[13] = f2bf(f3.y); f.us[14] = f2bf(f3.z); f.us[15] = f2bf(f3.w);
  return f.v;
}

__device__ __forceinline__ v8f wmma_bf16(v16bf a, v16bf b, v8f c) {
  // D = A * B + C ; emits v_wmma_f32_16x16x32_bf16
  return __builtin_amdgcn_wmma_f32_16x16x32_bf16(false, a, false, b, (short)0, c,
                                                 false, false);
}

__device__ __forceinline__ v8f zero8() {
  v8f z = {0.f, 0.f, 0.f, 0.f, 0.f, 0.f, 0.f, 0.f};
  return z;
}

// ---------------------------------------------------------------------------
// Weight repack: fp32 -> bf16 in exact WMMA B-fragment lane layout.
// Element mapping inside a fragment: idx = lane*16 + q ;
// n = lane&15 ; h = lane>>4 ; k_local = 8h + (q&7) + ((q&8)?16:0)
// ---------------------------------------------------------------------------
__global__ void crossnet_pack_kernel(const float* __restrict__ U,
                                     const float* __restrict__ V,
                                     const float* __restrict__ C,
                                     const float* __restrict__ gw,
                                     unsigned short* __restrict__ ws) {
  int tid = blockIdx.x * blockDim.x + threadIdx.x;
  if (tid >= TOTE) return;
  int q    = tid & 15;
  int lane = (tid >> 4) & 31;
  int n16  = lane & 15;
  int h    = lane >> 4;
  int klocal = 8 * h + (q & 7) + ((q & 8) ? 16 : 0);
  float val = 0.f;
  if (tid < VOFF) {
    // gates: B[k=d][n=e], zero-padded to 16 cols
    int kc = tid >> 9;
    int k = kc * 32 + klocal;
    val = (n16 < NEXP) ? gw[n16 * DIM + k] : 0.f;
  } else if (tid < COFF) {
    // Vcat: B[k=d][n = e*64 + r] = V[l,e,r,d]
    int t = tid - VOFF;
    int frag = t >> 9;                 // ((l*32 + kc)*16 + nt)
    int nt = frag & 15;
    int kc = (frag >> 4) & 31;
    int l  = frag >> 9;
    int d  = kc * 32 + klocal;
    int ncol = nt * 16 + n16;
    int e = ncol >> 6, r = ncol & 63;
    val = V[((size_t)((l * NEXP + e) * RANK + r)) * DIM + d];
  } else if (tid < UOFF) {
    // C: per-expert B[k=s][n=r] = C[l,e,r,s]
    int t = tid - COFF;
    int frag = t >> 9;                 // ((l*4+e)*2 + kc)*4 + nt2
    int nt2 = frag & 3;
    int kc  = (frag >> 2) & 1;
    int e   = (frag >> 3) & 3;
    int l   = frag >> 5;
    int s = kc * 32 + klocal;
    int r = nt2 * 16 + n16;
    val = C[(((l * NEXP + e) * RANK + r) << 6) + s];
  } else {
    // Ucat: B[k = e*64 + r][n=d] = U[l,e,d,r]
    int t = tid - UOFF;
    int frag = t >> 9;                 // (l*8 + kc)*64 + nt
    int nt = frag & 63;
    int kc = (frag >> 6) & 7;
    int l  = frag >> 9;
    int k = kc * 32 + klocal;
    int e = k >> 6, r = k & 63;
    int d = nt * 16 + n16;
    val = U[((size_t)((l * NEXP + e) * DIM + d)) * RANK + r];
  }
  ws[tid] = f2bf(val);
}

// ---------------------------------------------------------------------------
// Main fused kernel. 4 waves / block, each wave owns 16 rows, all 3 layers.
// LDS per block: xl fp32 (4*64KB) + bf16 t1/t2 stage (4*8KB, in-place reuse)
// + gate buffer (4*1KB) = 299008 B (dynamic, CDNA5 320KB/WGP).
// ---------------------------------------------------------------------------
#define WAVES 4
#define XL_F  (16 * DIM)    // floats per wave
#define STG_H (16 * 256)    // bf16 halfs per wave
#define GB_F  (16 * 16)     // floats per wave

__global__ __launch_bounds__(128, 1)
void crossnet_main_kernel(const float* __restrict__ x,
                          const float* __restrict__ bias,
                          const unsigned short* __restrict__ ws,
                          float* __restrict__ out) {
  extern __shared__ char smem[];
  const int wave = threadIdx.x >> 5;
  const int lane = threadIdx.x & 31;

  float*          xl   = reinterpret_cast<float*>(smem) + wave * XL_F;
  unsigned short* stg  = reinterpret_cast<unsigned short*>(smem + WAVES * XL_F * 4) + wave * STG_H;
  float*          gbuf = reinterpret_cast<float*>(smem + WAVES * XL_F * 4 + WAVES * STG_H * 2) + wave * GB_F;

  const int r0w = blockIdx.x * (WAVES * 16) + wave * 16;  // wave's first row

  // ---- init: xl <- x tile (coalesced) --------------------------------------
  {
    const float* src = x + (size_t)r0w * DIM;
    for (int idx = lane; idx < XL_F; idx += 32) xl[idx] = src[idx];
  }

  const unsigned short* gpack = ws + GOFF;

  for (int l = 0; l < LAYERS; ++l) {
    const unsigned short* vpackL = ws + VOFF + l * (V_FRAGS_PER_L * FRAG_E);
    const unsigned short* cpackL = ws + COFF + l * (C_FRAGS_PER_L * FRAG_E);
    const unsigned short* upackL = ws + UOFF + l * (U_FRAGS_PER_L * FRAG_E);
    const float* biasL = bias + l * DIM;

    const int col = lane & 15;
    const int h   = lane >> 4;

    // ---- Phase A+B: gate logits + T1 = XL @ Vcat, shared A over K=1024 ----
    v8f accV[16];
#pragma unroll
    for (int t = 0; t < 16; ++t) accV[t] = zero8();
    v8f accG = zero8();

#pragma unroll 1
    for (int kc = 0; kc < 32; ++kc) {
      v16bf a = afrag_f32(xl, DIM, kc * 32, lane);
      v16bf bg = load_bfrag(gpack + kc * FRAG_E, lane);
      accG = wmma_bf16(a, bg, accG);
#pragma unroll
      for (int nt = 0; nt < 16; ++nt) {
        v16bf b = load_bfrag(vpackL + (size_t)(kc * 16 + nt) * FRAG_E, lane);
        accV[nt] = wmma_bf16(a, b, accV[nt]);
      }
    }

    // dump gate logits to LDS, then per-row softmax over the 4 experts
#pragma unroll
    for (int v = 0; v < 8; ++v) gbuf[(v + 8 * h) * 16 + col] = accG[v];
    if (lane < 16) {
      float a0 = gbuf[lane * 16 + 0];
      float a1 = gbuf[lane * 16 + 1];
      float a2 = gbuf[lane * 16 + 2];
      float a3 = gbuf[lane * 16 + 3];
      float mx = fmaxf(fmaxf(a0, a1), fmaxf(a2, a3));
      float e0 = __expf(a0 - mx), e1 = __expf(a1 - mx);
      float e2 = __expf(a2 - mx), e3 = __expf(a3 - mx);
      float inv = 1.0f / (e0 + e1 + e2 + e3);
      gbuf[lane * 16 + 0] = e0 * inv;
      gbuf[lane * 16 + 1] = e1 * inv;
      gbuf[lane * 16 + 2] = e2 * inv;
      gbuf[lane * 16 + 3] = e3 * inv;
    }

    // tanh(T1) -> bf16 stage [16 rows][256 cols]
#pragma unroll
    for (int nt = 0; nt < 16; ++nt) {
#pragma unroll
      for (int v = 0; v < 8; ++v) {
        int row = v + 8 * h;
        stg[row * 256 + nt * 16 + col] = f2bf(tanhf(accV[nt][v]));
      }
    }

    // ---- Phase C: per-expert T2 = g_e * tanh(T1_e @ C_e^T), in-place ------
#pragma unroll 1
    for (int e = 0; e < NEXP; ++e) {
      v16bf a0 = afrag_bf16(stg, 256, e * 64, lane);
      v16bf a1 = afrag_bf16(stg, 256, e * 64 + 32, lane);
      v8f accC[4];
#pragma unroll
      for (int t = 0; t < 4; ++t) accC[t] = zero8();
#pragma unroll
      for (int nt2 = 0; nt2 < 4; ++nt2) {
        v16bf b0 = load_bfrag(cpackL + ((e * 2 + 0) * 4 + nt2) * FRAG_E, lane);
        v16bf b1 = load_bfrag(cpackL + ((e * 2 + 1) * 4 + nt2) * FRAG_E, lane);
        accC[nt2] = wmma_bf16(a0, b0, accC[nt2]);
        accC[nt2] = wmma_bf16(a1, b1, accC[nt2]);
      }
#pragma unroll
      for (int nt2 = 0; nt2 < 4; ++nt2) {
#pragma unroll
        for (int v = 0; v < 8; ++v) {
          int row = v + 8 * h;
          float gv = gbuf[row * 16 + e];
          stg[row * 256 + e * 64 + nt2 * 16 + col] = f2bf(tanhf(accC[nt2][v]) * gv);
        }
      }
    }

    // ---- Phase D: xl += x0 .* (S @ Ucat + bias) ---------------------------
    v16bf aU[8];
#pragma unroll
    for (int kc = 0; kc < 8; ++kc) aU[kc] = afrag_bf16(stg, 256, kc * 32, lane);

#pragma unroll 1
    for (int nt = 0; nt < 64; ++nt) {
      v8f acc = zero8();
#pragma unroll
      for (int kc = 0; kc < 8; ++kc) {
        v16bf b = load_bfrag(upackL + (size_t)(kc * 64 + nt) * FRAG_E, lane);
        acc = wmma_bf16(aU[kc], b, acc);
      }
#pragma unroll
      for (int v = 0; v < 8; ++v) {
        int row = v + 8 * h;
        int d = nt * 16 + col;
        float xlv = xl[row * DIM + d];
        float x0v = x[(size_t)(r0w + row) * DIM + d];
        float bi  = biasL[d];
        xl[row * DIM + d] = xlv + x0v * (acc[v] + bi);
      }
    }
  }

  // ---- write out (coalesced) ----------------------------------------------
  {
    float* dst = out + (size_t)r0w * DIM;
    for (int idx = lane; idx < XL_F; idx += 32) dst[idx] = xl[idx];
  }
}

// ---------------------------------------------------------------------------
extern "C" void kernel_launch(void* const* d_in, const int* in_sizes, int n_in,
                              void* d_out, int out_size, void* d_ws, size_t ws_size,
                              hipStream_t stream) {
  (void)in_sizes; (void)n_in; (void)out_size; (void)ws_size;
  const float* x    = (const float*)d_in[0];   // [B, D]
  const float* U    = (const float*)d_in[1];   // [L, E, D, R]
  const float* V    = (const float*)d_in[2];   // [L, E, R, D]
  const float* C    = (const float*)d_in[3];   // [L, E, R, R]
  const float* bias = (const float*)d_in[4];   // [L, D]
  const float* gw   = (const float*)d_in[5];   // [E, D]
  unsigned short* ws = (unsigned short*)d_ws;  // bf16 packed weights (~3.3 MB)
  float* out = (float*)d_out;

  // 1) repack weights fp32 -> bf16 WMMA fragments (small, L2-resident)
  crossnet_pack_kernel<<<(TOTE + 255) / 256, 256, 0, stream>>>(U, V, C, gw, ws);

  // 2) fused 3-layer CrossNetMix
  const size_t shmem = (size_t)WAVES * XL_F * 4   // xl fp32
                     + (size_t)WAVES * STG_H * 2  // t1/t2 bf16 stage
                     + (size_t)WAVES * GB_F * 4;  // gate buffer
  const int blocks = BATCH / (WAVES * 16);        // 256
  crossnet_main_kernel<<<blocks, 128, shmem, stream>>>(x, bias, ws, out);
}
